// Predictor_2396591751762
// MI455X (gfx1250) — matrix operand order repaired and checked by reading, not compiled, not run on hardware
//
#include <hip/hip_runtime.h>
#include <hip/hip_bf16.h>

typedef __attribute__((ext_vector_type(16))) __bf16         v16bf;
typedef __attribute__((ext_vector_type(16))) unsigned short v16u;
typedef __attribute__((ext_vector_type(8)))  float          v8f;

#define N_NODES 50000
#define N_EDGES 800000
// M tiles: 50000 / 16 = 3125 exactly

__device__ __forceinline__ unsigned short f2bf(float f) {
    unsigned u = __float_as_uint(f);
    u += 0x7FFFu + ((u >> 16) & 1u);          // round-to-nearest-even
    return (unsigned short)(u >> 16);
}
__device__ __forceinline__ float bf2f(unsigned short h) {
    return __uint_as_float(((unsigned)h) << 16);
}

__global__ void k_zero(float* __restrict__ p, int n) {
    int i = blockIdx.x * 256 + threadIdx.x;
    if (i < n) p[i] = 0.0f;
}

// Pack W2 [256,256] f32 -> bf16 in WMMA B-matrix (32x16 tile) lane layout.
// tile (kt,nt): lane = nn + 16*((kk/8)%2), element e = kk%8 + 8*(kk/16).
__global__ void k_packB(const float* __restrict__ W2, unsigned short* __restrict__ Bp) {
    int tid = blockIdx.x * 256 + threadIdx.x;      // 65536
    int k = tid >> 8, n = tid & 255;
    int kt = k >> 5, kk = k & 31;
    int half = (kk >> 3) & 1, e = (kk & 7) + ((kk >> 4) << 3);
    int nt = n >> 4, nn = n & 15;
    int lane = nn + (half << 4);
    Bp[(((kt << 4) + nt) * 32 + lane) * 16 + e] = f2bf(W2[tid]);
}

// h0[n] = (in_deg, out_deg)
__global__ void k_deg(const int* __restrict__ src, const int* __restrict__ dst,
                      float* __restrict__ h0) {
    int e = blockIdx.x * 256 + threadIdx.x;
    if (e < N_EDGES) {
        atomicAdd(&h0[dst[e] * 2 + 0], 1.0f);
        atomicAdd(&h0[src[e] * 2 + 1], 1.0f);
    }
}

__global__ void k_agg1(const int* __restrict__ src, const int* __restrict__ dst,
                       const float* __restrict__ h0, float* __restrict__ agg1) {
    int e = blockIdx.x * 256 + threadIdx.x;
    if (e < N_EDGES) {
        int s = src[e], d = dst[e];
        atomicAdd(&agg1[d * 2 + 0], h0[s * 2 + 0]);
        atomicAdd(&agg1[d * 2 + 1], h0[s * 2 + 1]);
    }
}

// h1 = relu(agg1 @ W1 + b1), stored bf16 (halves gather traffic of the hot stage)
__global__ void k_l1(const float* __restrict__ agg1, const float* __restrict__ W1,
                     const float* __restrict__ b1, unsigned short* __restrict__ h1) {
    int n = blockIdx.x, j = threadIdx.x;
    float a0 = agg1[n * 2 + 0], a1 = agg1[n * 2 + 1];
    float v = fmaf(a0, W1[j], fmaf(a1, W1[256 + j], b1[j]));
    h1[(size_t)n * 256 + j] = f2bf(fmaxf(v, 0.0f));
}

// agg2[dst] += h1[src]  (dominant stage: 64 threads per edge, 4 channels each)
__global__ void k_agg2(const int* __restrict__ src, const int* __restrict__ dst,
                       const unsigned short* __restrict__ h1, float* __restrict__ agg2) {
    long long gid = (long long)blockIdx.x * 256 + threadIdx.x;
    int e = (int)(gid >> 6);
    int c = ((int)gid & 63) * 4;
    if (e < N_EDGES) {
        int s = src[e], d = dst[e];
        ushort4 h4 = *(const ushort4*)(h1 + (size_t)s * 256 + c);
        float* ap = agg2 + (size_t)d * 256 + c;
        atomicAdd(ap + 0, bf2f(h4.x));
        atomicAdd(ap + 1, bf2f(h4.y));
        atomicAdd(ap + 2, bf2f(h4.z));
        atomicAdd(ap + 3, bf2f(h4.w));
    }
}

// WMMA GEMM: h2 = agg2[50000,256] @ W2[256,256] + b2 ; ge += colsum(relu(h2))
// One wave per 16-row tile; A row-block held in 64 VGPRs; B preloaded in lane layout.
__global__ __launch_bounds__(128) void k_gemm2(const float* __restrict__ agg2,
                                               const unsigned short* __restrict__ Bp,
                                               const float* __restrict__ b2,
                                               float* __restrict__ ge) {
    __shared__ float lds[256];
    int tid = threadIdx.x;
    for (int i = tid; i < 256; i += 128) lds[i] = 0.0f;
    __syncthreads();

    int wave = tid >> 5, lane = tid & 31;
    int tileM = blockIdx.x * 4 + wave;
    if (tileM < 3125) {                                   // wave-uniform branch
        int half = lane >> 4, m = lane & 15;
        const float* rp = agg2 + (size_t)(tileM * 16 + m) * 256;

        // A fragments: 16-bit A 16x32 layout: elems 0..7 -> K = kb..kb+7,
        // elems 8..15 -> K = kb+16..kb+23, kb = 32*kt + 8*half
        v16bf a[8];
#pragma unroll
        for (int kt = 0; kt < 8; ++kt) {
            int kb = kt * 32 + half * 8;
            v16u au;
#pragma unroll
            for (int e = 0; e < 8; ++e) au[e] = f2bf(rp[kb + e]);
#pragma unroll
            for (int e = 0; e < 8; ++e) au[8 + e] = f2bf(rp[kb + 16 + e]);
            a[kt] = __builtin_bit_cast(v16bf, au);
        }

        const v16u* Bt = (const v16u*)Bp;
        int n = lane & 15;
#pragma unroll 1
        for (int nt = 0; nt < 16; ++nt) {
            v8f acc = {};
#pragma unroll
            for (int kt = 0; kt < 8; ++kt) {
                v16bf b = __builtin_bit_cast(v16bf, Bt[(kt * 16 + nt) * 32 + lane]);
                acc = __builtin_amdgcn_wmma_f32_16x16x32_bf16(
                    false, a[kt], false, b, (short)0, acc, false, false);
            }
            // epilogue: bias + relu + column sum (M-reduction)
            float bias = b2[nt * 16 + n];
            float s = 0.0f;
#pragma unroll
            for (int r = 0; r < 8; ++r) s += fmaxf(acc[r] + bias, 0.0f);
            s += __shfl_xor(s, 16, 32);                   // merge M halves
            if (lane < 16) atomicAdd(&lds[nt * 16 + n], s);
        }
    }
    __syncthreads();
    for (int i = tid; i < 256; i += 128) atomicAdd(&ge[i], lds[i]);
}

// params MLP + predictor MLP, single block
__global__ __launch_bounds__(256) void k_head(float* __restrict__ out,
        const float* __restrict__ params,
        const float* __restrict__ pW1, const float* __restrict__ pb1,
        const float* __restrict__ pW2, const float* __restrict__ pb2,
        const float* __restrict__ qW1, const float* __restrict__ qb1,
        const float* __restrict__ qW2, const float* __restrict__ qb2) {
    __shared__ float ph[64], pe[64], zh[256], ge[256];
    int t = threadIdx.x;
    ge[t] = out[t];
    if (t < 64) {
        float s = pb1[t];
        for (int i = 0; i < 16; ++i) s = fmaf(params[i], pW1[i * 64 + t], s);
        ph[t] = fmaxf(s, 0.0f);
    }
    __syncthreads();
    if (t < 64) {
        float s = pb2[t];
        for (int i = 0; i < 64; ++i) s = fmaf(ph[i], pW2[i * 64 + t], s);
        pe[t] = s;
        out[256 + t] = s;
    }
    __syncthreads();
    {
        float s = qb1[t];
        for (int i = 0; i < 256; ++i) s = fmaf(ge[i], qW1[i * 256 + t], s);
        for (int i = 0; i < 64; ++i)  s = fmaf(pe[i], qW1[(256 + i) * 256 + t], s);
        zh[t] = fmaxf(s, 0.0f);
    }
    __syncthreads();
    if (t < 4) {
        float s = qb2[t];
        for (int i = 0; i < 256; ++i) s = fmaf(zh[i], qW2[i * 4 + t], s);
        out[320 + t] = s;
    }
}

extern "C" void kernel_launch(void* const* d_in, const int* in_sizes, int n_in,
                              void* d_out, int out_size, void* d_ws, size_t ws_size,
                              hipStream_t stream) {
    const int*   src    = (const int*)d_in[0];
    const int*   dst    = (const int*)d_in[1];
    const float* params = (const float*)d_in[2];
    const float* W1     = (const float*)d_in[3];
    const float* b1     = (const float*)d_in[4];
    const float* W2     = (const float*)d_in[5];
    const float* b2     = (const float*)d_in[6];
    const float* pW1    = (const float*)d_in[7];
    const float* pb1    = (const float*)d_in[8];
    const float* pW2    = (const float*)d_in[9];
    const float* pb2    = (const float*)d_in[10];
    const float* qW1    = (const float*)d_in[11];
    const float* qb1    = (const float*)d_in[12];
    const float* qW2    = (const float*)d_in[13];
    const float* qb2    = (const float*)d_in[14];
    float* out = (float*)d_out;

    char* ws = (char*)d_ws;
    float*          h0   = (float*)(ws + 0);                    // 100000 f32
    float*          agg1 = (float*)(ws + 400384);               // 100000 f32
    float*          agg2 = (float*)(ws + 800768);               // 12,800,000 f32
    unsigned short* h1   = (unsigned short*)(ws + 52000768);    // 12,800,000 bf16
    unsigned short* Bp   = (unsigned short*)(ws + 77600768);    // 65536 bf16

    k_zero <<<(100000   + 255) / 256, 256, 0, stream>>>(h0,   100000);
    k_zero <<<(100000   + 255) / 256, 256, 0, stream>>>(agg1, 100000);
    k_zero <<<(12800000 + 255) / 256, 256, 0, stream>>>(agg2, 12800000);
    k_zero <<<2, 256, 0, stream>>>(out, 324);
    k_packB<<<256, 256, 0, stream>>>(W2, Bp);
    k_deg  <<<3125, 256, 0, stream>>>(src, dst, h0);
    k_agg1 <<<3125, 256, 0, stream>>>(src, dst, h0, agg1);
    k_l1   <<<50000, 256, 0, stream>>>(agg1, W1, b1, h1);
    k_agg2 <<<200000, 256, 0, stream>>>(src, dst, h1, agg2);
    k_gemm2<<<782, 128, 0, stream>>>(agg2, Bp, b2, out);
    k_head <<<1, 256, 0, stream>>>(out, params, pW1, pb1, pW2, pb2,
                                   qW1, qb1, qW2, qb2);
}
